// SelfAttention_42838003810533
// MI455X (gfx1250) — matrix operand order repaired and checked by reading, not compiled, not run on hardware
//
#include <hip/hip_runtime.h>

typedef __bf16 bf16_t;
typedef __attribute__((ext_vector_type(16))) __bf16 v16bf;
typedef __attribute__((ext_vector_type(8)))  __bf16 v8bf;
typedef __attribute__((ext_vector_type(8)))  float  v8f;

typedef unsigned int u32;
typedef unsigned long long u64;
typedef __attribute__((ext_vector_type(4))) unsigned int u32x4;
typedef __attribute__((ext_vector_type(8))) int i32x8;
typedef __attribute__((ext_vector_type(4))) int i32x4;

#define NB   4
#define CIN  256
#define NSP  4096
#define DAT  64
#define OTOT 384   // 64 q + 64 k + 256 v output channels

#if __has_builtin(__builtin_amdgcn_tensor_load_to_lds) && \
    __has_builtin(__builtin_amdgcn_s_wait_tensorcnt)
#define HAS_TDM 1
#else
#define HAS_TDM 0
#endif

__device__ __forceinline__ v8f wmma_bf16(v16bf a, v16bf b, v8f c) {
  // D = A(16x32) * B(32x16) + C, f32 accumulate
  return __builtin_amdgcn_wmma_f32_16x16x32_bf16(
      false, a, false, b, (short)0, c, false, false);
}

// A fragment (16x32, bf16): tile at `base`, row-major, `stride` elements/row.
// ISA 7.12.2: lane m=l&15; VGPR0..3 hold k=hi*8+{0..7}, VGPR4..7 hold 16+hi*8+{0..7}.
__device__ __forceinline__ v16bf frag_A(const bf16_t* base, int stride, int lane) {
  int m  = lane & 15;
  int hi = (lane >> 4) & 1;
  const bf16_t* p = base + (size_t)m * stride + hi * 8;
  v8bf lo = *(const v8bf*)(p);
  v8bf up = *(const v8bf*)(p + 16);
  v16bf r;
#pragma unroll
  for (int i = 0; i < 8; ++i) { r[i] = lo[i]; r[i + 8] = up[i]; }
  return r;
}

// B fragment (32x16, bf16): B[k][n] where row-major storage M2[n][k] is at `base`.
__device__ __forceinline__ v16bf frag_B(const bf16_t* base, int stride, int lane) {
  int n  = lane & 15;
  int hi = (lane >> 4) & 1;
  const bf16_t* p = base + (size_t)n * stride + hi * 8;
  v8bf lo = *(const v8bf*)(p);
  v8bf up = *(const v8bf*)(p + 16);
  v16bf r;
#pragma unroll
  for (int i = 0; i < 8; ++i) { r[i] = lo[i]; r[i + 8] = up[i]; }
  return r;
}

#if HAS_TDM
// Issue one TDM 2-D tile load (bf16 elements) global -> LDS.
// Tensor: dim0 (contiguous) x dim1, row stride `stride0` elements.
// Tile: tile0 x tile1 starting at `gaddr`, landing contiguous at `lds_off`.
// D# layout per CDNA5 ISA 8.3/8.4 (group0: count/lds/global/type, group1: dims).
// This toolchain exposes the 6-arg builtin: (g0, g1, g2, g3, g_extra, cpol).
__device__ __forceinline__ void tdm_load_2d(const bf16_t* gaddr, u32 lds_off,
                                            u32 dim0, u32 dim1,
                                            u32 tile0, u32 tile1, u32 stride0) {
  u64 ga = (u64)(size_t)gaddr;
  u32x4 g0;
  g0[0] = 1u;                              // count=1, user descriptor
  g0[1] = lds_off;                         // lds_addr (bytes)
  g0[2] = (u32)ga;                         // global_addr[31:0]
  g0[3] = (u32)(ga >> 32) | (2u << 30);    // global_addr[56:32] | type=2
  i32x8 g1;
  g1[0] = (int)(1u << 16);                 // data_size=1 -> 2-byte elements
  g1[1] = (int)(dim0 << 16);               // tensor_dim0[15:0]
  g1[2] = (int)((dim0 >> 16) | (dim1 << 16));
  g1[3] = (int)((dim1 >> 16) | (tile0 << 16));
  g1[4] = (int)(tile1 & 0xffffu);          // tile_dim1 (tile_dim2 = 0)
  g1[5] = (int)stride0;                    // tensor_dim0_stride[31:0]
  g1[6] = 0;                               // stride[47:32] | dim1_stride lo (2D: unused)
  g1[7] = 0;
  i32x4 z4 = {0, 0, 0, 0};                 // groups 2/3: unused for 2-D tensor
  i32x8 z8 = {0, 0, 0, 0, 0, 0, 0, 0};     // extra group: unused
  __builtin_amdgcn_tensor_load_to_lds(g0, g1, z4, z4, z8, 0);
}
#endif

// ---------------------------------------------------------------------------
// Kernel 1: fused QKV projection.  y[n,o] = sum_c W[o,c] * x[b,c,n] + bias[o]
// o in [0,64)->Q[N,64], [64,128)->K[N,64], [128,384)->V^T stored [256,N].
// ---------------------------------------------------------------------------
__global__ __launch_bounds__(128) void qkv_proj_kernel(
    const float* __restrict__ x,
    const float* __restrict__ wq, const float* __restrict__ bq,
    const float* __restrict__ wk, const float* __restrict__ bk,
    const float* __restrict__ wv, const float* __restrict__ bv,
    bf16_t* __restrict__ qout,   // [B, N, 64]
    bf16_t* __restrict__ kout,   // [B, N, 64]
    bf16_t* __restrict__ vout)   // [B, 256, N]  (transposed)
{
  const int tile = blockIdx.x;
  const int b  = tile / ((NSP / 16) * (OTOT / 64));
  const int r  = tile % ((NSP / 16) * (OTOT / 64));
  const int n0 = (r / (OTOT / 64)) * 16;
  const int o0 = (r % (OTOT / 64)) * 64;
  const int tid  = threadIdx.x;
  const int lane = tid & 31;
  const int wave = tid >> 5;

  __shared__ bf16_t xs[16][32];   // A tile: [n_local][c_local]
  __shared__ bf16_t wsm[64][32];  // B tile (transposed view): [o_local][c_local]

  v8f acc = {};
  const float* xb = x + (size_t)b * CIN * NSP;

  for (int c0 = 0; c0 < CIN; c0 += 32) {
    {
      int nn = tid & 15;
      int cb = tid >> 4;
#pragma unroll
      for (int i = 0; i < 4; ++i) {
        int cc = cb + i * 8;
        xs[nn][cc] = (bf16_t)xb[(size_t)(c0 + cc) * NSP + n0 + nn];
      }
    }
    {
      int cl = (tid & 7) * 4;
      int ob = tid >> 3;
#pragma unroll
      for (int i = 0; i < 4; ++i) {
        int ol = ob + i * 16;
        int og = o0 + ol;
        const float* wrow = (og < 64)  ? wq + (size_t)og * CIN
                          : (og < 128) ? wk + (size_t)(og - 64) * CIN
                                       : wv + (size_t)(og - 128) * CIN;
#pragma unroll
        for (int jj = 0; jj < 4; ++jj)
          wsm[ol][cl + jj] = (bf16_t)wrow[c0 + cl + jj];
      }
    }
    __syncthreads();
    v16bf a  = frag_A(&xs[0][0], 32, lane);
    v16bf bb = frag_B(&wsm[wave * 16][0], 32, lane);
    acc = wmma_bf16(a, bb, acc);
    __syncthreads();
  }

  const int oc = o0 + wave * 16 + (lane & 15);
  const int hi = (lane >> 4) & 1;
  const float bias = (oc < 64) ? bq[oc] : (oc < 128) ? bk[oc - 64] : bv[oc - 128];
#pragma unroll
  for (int j = 0; j < 8; ++j) {
    int n = n0 + j + hi * 8;
    bf16_t h = (bf16_t)(acc[j] + bias);
    if (oc < 64)        qout[((size_t)b * NSP + n) * DAT + oc] = h;
    else if (oc < 128)  kout[((size_t)b * NSP + n) * DAT + (oc - 64)] = h;
    else                vout[((size_t)(b * CIN + (oc - 128))) * NSP + n] = h;
  }
}

// ---------------------------------------------------------------------------
// Kernel 2: flash attention, TDM double-buffered.
// 8 waves x 16-query tiles; K tile (32x64) + V tile (256x32, [o][key]) staged
// in LDS. With TDM, wave0 issues the next block's two tensor loads, waits
// tensorcnt<=2 (current block complete, in-order), and compute overlaps DMA.
// ---------------------------------------------------------------------------
__global__ __launch_bounds__(256) void flash_attn_kernel(
    const bf16_t* __restrict__ qm,  // [B, N, 64]
    const bf16_t* __restrict__ km,  // [B, N, 64]
    const bf16_t* __restrict__ vm,  // [B, 256, N]
    float* __restrict__ out)        // [B, 256, N]
{
  const int tid  = threadIdx.x;
  const int lane = tid & 31;
  const int wave = tid >> 5;
  const int qt = blockIdx.x * 8 + wave;
  const int b  = qt >> 8;                // uniform across the block
  const int n0 = (qt & 255) * 16;
  const int hi = (lane >> 4) & 1;

  __shared__ bf16_t ks[2][32][64];       // K tiles  [buf][key_local][d]
  __shared__ bf16_t vs[2][256][32];      // V tiles  [buf][o][key_local]
  __shared__ bf16_t plds[8][16 * 32];    // per-wave P staging (C->A layout)
  bf16_t* pw = &plds[wave][0];

  const bf16_t* qb = qm + ((size_t)b * NSP + n0) * DAT;
  const bf16_t* kb = km + (size_t)b * NSP * DAT;
  const bf16_t* vb = vm + (size_t)b * CIN * NSP;

  const v16bf aq0 = frag_A(qb, DAT, lane);
  const v16bf aq1 = frag_A(qb + 32, DAT, lane);

  float mrow[8], lrow[8];
  v8f acc[16];
#pragma unroll
  for (int j = 0; j < 8; ++j) { mrow[j] = -__builtin_inff(); lrow[j] = 0.f; }
#pragma unroll
  for (int t = 0; t < 16; ++t) acc[t] = (v8f){};

#if HAS_TDM
  if (wave == 0) {  // prologue: start DMA of block 0 into buffer 0
    tdm_load_2d(kb, (u32)(size_t)&ks[0][0][0], DAT, NSP, DAT, 32, DAT);
    tdm_load_2d(vb, (u32)(size_t)&vs[0][0][0], NSP, CIN, 32, CIN, NSP);
  }
#endif

  for (int k0 = 0; k0 < NSP; k0 += 32) {
    const int cur = (k0 >> 5) & 1;
#if HAS_TDM
    if (wave == 0) {
      if (k0 + 32 < NSP) {  // kick off next block into the other buffer
        const int nxt = cur ^ 1;
        tdm_load_2d(kb + (size_t)(k0 + 32) * DAT,
                    (u32)(size_t)&ks[nxt][0][0], DAT, NSP, DAT, 32, DAT);
        tdm_load_2d(vb + (k0 + 32),
                    (u32)(size_t)&vs[nxt][0][0], NSP, CIN, 32, CIN, NSP);
        __builtin_amdgcn_s_wait_tensorcnt((short)2);  // current block done
      } else {
        __builtin_amdgcn_s_wait_tensorcnt((short)0);
      }
    }
    __syncthreads();
#else
    {
      int krow = tid >> 3;
      int kc   = (tid & 7) * 8;
      *(v8bf*)&ks[cur][krow][kc] =
          *(const v8bf*)(kb + (size_t)(k0 + krow) * DAT + kc);
      const bf16_t* src = vb + (size_t)tid * NSP + k0;
      *(v16bf*)&vs[cur][tid][0]  = *(const v16bf*)(src);
      *(v16bf*)&vs[cur][tid][16] = *(const v16bf*)(src + 16);
      if (k0 + 32 < NSP) __builtin_prefetch(src + 32, 0, 1);
    }
    __syncthreads();
#endif

    // ---- scores: two 16x16 tiles over 32 keys, reduced over d=64 ----
    v8f s0 = {}, s1 = {};
    s0 = wmma_bf16(aq0, frag_B(&ks[cur][0][0], DAT, lane), s0);
    s0 = wmma_bf16(aq1, frag_B(&ks[cur][0][0] + 32, DAT, lane), s0);
    s1 = wmma_bf16(aq0, frag_B(&ks[cur][16][0], DAT, lane), s1);
    s1 = wmma_bf16(aq1, frag_B(&ks[cur][16][0] + 32, DAT, lane), s1);

    // ---- online softmax: row stats live in 16-lane halves (C/D layout) ----
#pragma unroll
    for (int j = 0; j < 8; ++j) {
      float t = fmaxf(s0[j], s1[j]);
      t = fmaxf(t, __shfl_xor(t, 1, 32));
      t = fmaxf(t, __shfl_xor(t, 2, 32));
      t = fmaxf(t, __shfl_xor(t, 4, 32));
      t = fmaxf(t, __shfl_xor(t, 8, 32));
      float mn = fmaxf(mrow[j], t);
      float sc = __expf(mrow[j] - mn);
      mrow[j] = mn;
      float p0 = __expf(s0[j] - mn);
      float p1 = __expf(s1[j] - mn);
      float rs = p0 + p1;
      rs += __shfl_xor(rs, 1, 32);
      rs += __shfl_xor(rs, 2, 32);
      rs += __shfl_xor(rs, 4, 32);
      rs += __shfl_xor(rs, 8, 32);
      lrow[j] = lrow[j] * sc + rs;
#pragma unroll
      for (int t2 = 0; t2 < 16; ++t2) acc[t2][j] *= sc;
      int row = j + hi * 8;
      pw[row * 32 + (lane & 15)]      = (bf16_t)p0;
      pw[row * 32 + (lane & 15) + 16] = (bf16_t)p1;
    }
#if __has_builtin(__builtin_amdgcn_s_wait_dscnt)
    __builtin_amdgcn_s_wait_dscnt(0);
#endif
    const v16bf ap = frag_A(pw, 32, lane);

    // ---- P*V: 16 o-tiles of 16 channels, operands straight from LDS ----
#pragma unroll
    for (int t = 0; t < 16; ++t) {
      v16bf bv = frag_B(&vs[cur][t * 16][0], 32, lane);
      acc[t] = wmma_bf16(ap, bv, acc[t]);
    }
    __syncthreads();  // all waves done with buf[cur] before it is re-filled
  }

  // ---- normalize and store out[b,c,n] (f32); 8 contiguous n per lane ----
  float inv[8];
#pragma unroll
  for (int j = 0; j < 8; ++j) inv[j] = 1.0f / lrow[j];
#pragma unroll
  for (int t = 0; t < 16; ++t) {
    int c = t * 16 + (lane & 15);
    float* op = out + ((size_t)(b * CIN + c)) * NSP + n0 + hi * 8;
#pragma unroll
    for (int j = 0; j < 8; ++j) op[j] = acc[t][j] * inv[j];
  }
}

// ---------------------------------------------------------------------------
extern "C" void kernel_launch(void* const* d_in, const int* in_sizes, int n_in,
                              void* d_out, int out_size, void* d_ws, size_t ws_size,
                              hipStream_t stream) {
  (void)in_sizes; (void)n_in; (void)out_size; (void)ws_size;
  const float* x  = (const float*)d_in[0];
  const float* wq = (const float*)d_in[1];
  const float* bq = (const float*)d_in[2];
  const float* wk = (const float*)d_in[3];
  const float* bk = (const float*)d_in[4];
  const float* wv = (const float*)d_in[5];
  const float* bv = (const float*)d_in[6];
  float* out = (float*)d_out;

  bf16_t* qws = (bf16_t*)d_ws;                       // B*N*64 bf16
  bf16_t* kws = qws + (size_t)NB * NSP * DAT;        // B*N*64 bf16
  bf16_t* vws = kws + (size_t)NB * NSP * DAT;        // B*256*N bf16

  const int proj_blocks = NB * (NSP / 16) * (OTOT / 64);  // 6144
  qkv_proj_kernel<<<proj_blocks, 128, 0, stream>>>(x, wq, bq, wk, bk, wv, bv,
                                                   qws, kws, vws);

  const int attn_blocks = NB * (NSP / 16) / 8;            // 128
  flash_attn_kernel<<<attn_blocks, 256, 0, stream>>>(qws, kws, vws, out);
}
